// MoE_42133629174213
// MI455X (gfx1250) — matrix operand order repaired and checked by reading, not compiled, not run on hardware
//
#include <hip/hip_runtime.h>
#include <hip/hip_bf16.h>
#include <cstdint>

#define EMB   1024
#define HID   1024
#define NE    16
#define NTOK  8192   // 2 * 4096 tokens

typedef __attribute__((ext_vector_type(16))) __bf16 v16bf;
typedef __attribute__((ext_vector_type(8)))  float  v8f;

union FragBF { v16bf v; uint4 q[2]; };

// ---------- helpers ----------
__device__ __forceinline__ unsigned short f2bf(float f) {
  union { float f; unsigned int u; } c; c.f = f;
  unsigned int u = c.u;
  unsigned int r = u + 0x7FFFu + ((u >> 16) & 1u);   // round-to-nearest-even
  return (unsigned short)(r >> 16);
}

__device__ __forceinline__ v8f bfwmma(v16bf a, v16bf b, v8f c) {
  // (neg_a, A, neg_b, B, c_mod, C, reuse_a, reuse_b)
  return __builtin_amdgcn_wmma_f32_16x16x32_bf16(false, a, false, b, (short)0, c,
                                                 false, false);
}

// 16B global->LDS async copy, per-lane addresses both sides (ASYNCcnt tracked).
// LDS byte address = low 32 bits of the generic pointer (aperture: addr[31:0]).
__device__ __forceinline__ void async_cp16(void* lds_dst, const void* gsrc) {
  unsigned loff = (unsigned)(uintptr_t)lds_dst;
  unsigned long long ga = (unsigned long long)(uintptr_t)gsrc;
  asm volatile("global_load_async_to_lds_b128 %0, %1, off"
               :: "v"(loff), "v"(ga) : "memory");
}
__device__ __forceinline__ void wait_async0() {
  asm volatile("s_wait_asynccnt 0x0" ::: "memory");
}

// A-matrix bf16 16x32 fragment (ISA 7.12.2): lane L<16 -> row M=L, K=0..7,16..23;
// lane L>=16 -> row M=L-16, K=8..15,24..31.  LDS tile is [BM][40] row-major bf16.
__device__ __forceinline__ v16bf ldsA(const unsigned short* A, int mi, int lane) {
  int m = mi * 16 + (lane & 15);
  int g = lane >> 4;
  const unsigned short* p = A + m * 40 + g * 8;
  FragBF f;
  f.q[0] = *(const uint4*)p;                 // K = 8g .. 8g+7
  f.q[1] = *(const uint4*)(p + 16);          // K = 8g+16 .. 8g+23
  return f.v;
}

// B-matrix bf16 32x16 fragment: lane L holds column N=L%16, K=8g..8g+7,8g+16..8g+23 (g=L/16).
// LDS tile is [32][stride] row-major bf16 (K-major), n contiguous.
__device__ __forceinline__ v16bf ldsB(const unsigned short* B, int ni, int lane,
                                      int stride) {
  const unsigned short* p = B + ni * 16 + (lane & 15);
  int g = lane >> 4;
  FragBF f;
  unsigned short* fe = (unsigned short*)&f;
#pragma unroll
  for (int s = 0; s < 16; s++) {
    int k = (s < 8) ? (8 * g + s) : (16 + 8 * g + (s - 8));
    fe[s] = p[k * stride];
  }
  return f.v;
}

// pack 8 consecutive f32 at p into 8 bf16 (one uint4)
__device__ __forceinline__ uint4 cvt8(const float* p) {
  float4 f0 = *(const float4*)p, f1 = *(const float4*)(p + 4);
  uint4 u;
  u.x = (unsigned)f2bf(f0.x) | ((unsigned)f2bf(f0.y) << 16);
  u.y = (unsigned)f2bf(f0.z) | ((unsigned)f2bf(f0.w) << 16);
  u.z = (unsigned)f2bf(f1.x) | ((unsigned)f2bf(f1.y) << 16);
  u.w = (unsigned)f2bf(f1.z) | ((unsigned)f2bf(f1.w) << 16);
  return u;
}

// ---------- kernel 0: x -> bf16, zero out, zero counts ----------
__global__ void k_init(const float* __restrict__ x, unsigned short* __restrict__ xb,
                       float* __restrict__ out, int* __restrict__ counts) {
  int i = (blockIdx.x * blockDim.x + threadIdx.x) * 4;
  if (i < NTOK * EMB) {
    float4 xv = *(const float4*)(x + i);
    uint2 b;
    b.x = (unsigned int)f2bf(xv.x) | ((unsigned int)f2bf(xv.y) << 16);
    b.y = (unsigned int)f2bf(xv.z) | ((unsigned int)f2bf(xv.w) << 16);
    *(uint2*)(xb + i) = b;
    float4 z = make_float4(0.f, 0.f, 0.f, 0.f);
    *(float4*)(out + i) = z;
  }
  if (blockIdx.x == 0 && threadIdx.x < NE) counts[threadIdx.x] = 0;
}

// ---------- kernel 1: router (softmax + top-2, scatter to expert lists) ----------
__global__ void k_router(const float* __restrict__ x, const float* __restrict__ wg,
                         int* __restrict__ counts, int* __restrict__ tok_id,
                         float* __restrict__ tok_gate) {
  int t = blockIdx.x * blockDim.x + threadIdx.x;
  if (t >= NTOK) return;
  const float* xr = x + (long)t * EMB;
  float acc[NE];
#pragma unroll
  for (int e = 0; e < NE; e++) acc[e] = 0.f;
  for (int k = 0; k < EMB; k++) {
    float xv = xr[k];
    const float* w = wg + k * NE;
#pragma unroll
    for (int e = 0; e < NE; e++) acc[e] = fmaf(xv, w[e], acc[e]);
  }
  float mx = acc[0];
#pragma unroll
  for (int e = 1; e < NE; e++) mx = fmaxf(mx, acc[e]);
  float p[NE], sum = 0.f;
#pragma unroll
  for (int e = 0; e < NE; e++) { p[e] = __expf(acc[e] - mx); sum += p[e]; }
  float inv = 1.f / sum;
  int i1 = 0, i2 = 0; float v1 = -1.f, v2 = -1.f;
#pragma unroll
  for (int e = 0; e < NE; e++) {
    float pe = p[e] * inv;
    if (pe > v1) { v2 = v1; i2 = i1; v1 = pe; i1 = e; }
    else if (pe > v2) { v2 = pe; i2 = e; }
  }
  int s1 = atomicAdd(&counts[i1], 1);
  tok_id[i1 * NTOK + s1] = t; tok_gate[i1 * NTOK + s1] = v1;
  int s2 = atomicAdd(&counts[i2], 1);
  tok_id[i2 * NTOK + s2] = t; tok_gate[i2 * NTOK + s2] = v2;
}

// ---------- kernel 2: exclusive scan of 16 counts ----------
__global__ void k_offsets(const int* __restrict__ counts, int* __restrict__ offsets) {
  if (blockIdx.x == 0 && threadIdx.x == 0) {
    int r = 0;
    for (int e = 0; e < NE; e++) { offsets[e] = r; r += counts[e]; }
    offsets[NE] = r;
  }
}

// ---------- kernel 3: H = silu(X*W1) .* (X*W2); tile 64x128, 8 waves ----------
// wave wv: M-subtiles {mi0, mi0+1}, N-subtiles {ni0, ni0+1} of 4x8 grid
__global__ __launch_bounds__(256)
void k_gemm1(const unsigned short* __restrict__ xb, const float* __restrict__ w1,
             const float* __restrict__ w2, const int* __restrict__ counts,
             const int* __restrict__ offsets, const int* __restrict__ tok_id,
             unsigned short* __restrict__ H) {
  int e   = blockIdx.z;
  int cnt = counts[e];
  int tm  = blockIdx.y;
  if (tm * 64 >= cnt) return;          // block-uniform: EXEC all-1 at WMMA
  int bn  = blockIdx.x;

  __shared__ unsigned short Al[64 * 40];
  __shared__ unsigned short B1l[32 * 136];
  __shared__ unsigned short B2l[32 * 136];

  int tid  = threadIdx.x;
  // A staging: 64 rows x 32 bf16, one 16B async copy per thread; clamp OOB rows
  int arow = tid >> 2, aseg = (tid & 3) * 8;
  int srowA = tm * 64 + arow; if (srowA >= cnt) srowA = cnt - 1;
  const unsigned short* asrc0 =
      xb + (long)tok_id[e * NTOK + srowA] * EMB + aseg;
  unsigned short* adst = Al + arow * 40 + aseg;
  // B staging: 32 rows x 128 cols f32 -> bf16, 16 cols per thread
  int brow = tid >> 3, bcol = (tid & 7) * 16;
  const float* w1e = w1 + (long)e * EMB * HID + (long)brow * HID + bn * 128 + bcol;
  const float* w2e = w2 + (long)e * EMB * HID + (long)brow * HID + bn * 128 + bcol;

  int lane = tid & 31;
  int wv   = tid >> 5;
  int mi0  = (wv & 1) * 2;
  int ni0  = (wv >> 1) * 2;

  v8f p100 = {}; v8f p101 = {}; v8f p110 = {}; v8f p111 = {};
  v8f p200 = {}; v8f p201 = {}; v8f p210 = {}; v8f p211 = {};

  for (int k0 = 0; k0 < EMB; k0 += 32) {
    // ---- A tile: async global->LDS (bf16 bytes, no VALU) ----
    async_cp16(adst, asrc0 + k0);
    // ---- B1/B2 tiles: f32 -> bf16 ----
    {
      const float* p1 = w1e + (long)k0 * HID;
      const float* p2 = w2e + (long)k0 * HID;
      if (k0 + 32 < EMB) {
        __builtin_prefetch(p1 + 32 * HID, 0, 1);
        __builtin_prefetch(p2 + 32 * HID, 0, 1);
      }
      unsigned short* d1 = B1l + brow * 136 + bcol;
      unsigned short* d2 = B2l + brow * 136 + bcol;
      *(uint4*)d1       = cvt8(p1);
      *(uint4*)(d1 + 8) = cvt8(p1 + 8);
      *(uint4*)d2       = cvt8(p2);
      *(uint4*)(d2 + 8) = cvt8(p2 + 8);
    }
    wait_async0();
    __syncthreads();
    v16bf a0  = ldsA(Al, mi0,     lane);
    v16bf a1  = ldsA(Al, mi0 + 1, lane);
    v16bf b10 = ldsB(B1l, ni0,     lane, 136);
    v16bf b11 = ldsB(B1l, ni0 + 1, lane, 136);
    v16bf b20 = ldsB(B2l, ni0,     lane, 136);
    v16bf b21 = ldsB(B2l, ni0 + 1, lane, 136);
    p100 = bfwmma(a0, b10, p100);
    p101 = bfwmma(a0, b11, p101);
    p110 = bfwmma(a1, b10, p110);
    p111 = bfwmma(a1, b11, p111);
    p200 = bfwmma(a0, b20, p200);
    p201 = bfwmma(a0, b21, p201);
    p210 = bfwmma(a1, b20, p210);
    p211 = bfwmma(a1, b21, p211);
    __syncthreads();
  }

  // ---- SwiGLU epilogue, write bf16 H ----
  int hi   = lane >> 4;
  int col0 = bn * 128 + ni0 * 16 + (lane & 15);
  int col1 = col0 + 16;
  int offs = offsets[e];
#pragma unroll
  for (int m = 0; m < 2; m++) {
    v8f g0 = m ? p110 : p100, g1 = m ? p111 : p101;
    v8f u0 = m ? p210 : p200, u1 = m ? p211 : p201;
#pragma unroll
    for (int r = 0; r < 8; r++) {
      int srow = tm * 64 + (mi0 + m) * 16 + r + 8 * hi;
      if (srow < cnt) {
        float h0 = (g0[r] / (1.f + __expf(-g0[r]))) * u0[r];
        float h1 = (g1[r] / (1.f + __expf(-g1[r]))) * u1[r];
        long hr = (long)(offs + srow) * HID;
        H[hr + col0] = f2bf(h0);
        H[hr + col1] = f2bf(h1);
      }
    }
  }
}

// ---------- kernel 4: out += gate * (H @ Wc); tile 64x128 ----------
__global__ __launch_bounds__(256)
void k_gemm2(const unsigned short* __restrict__ H, const float* __restrict__ wc,
             const int* __restrict__ counts, const int* __restrict__ offsets,
             const int* __restrict__ tok_id, const float* __restrict__ tok_gate,
             float* __restrict__ out) {
  int e   = blockIdx.z;
  int cnt = counts[e];
  int tm  = blockIdx.y;
  if (tm * 64 >= cnt) return;
  int bn  = blockIdx.x;

  __shared__ unsigned short Al[64 * 40];
  __shared__ unsigned short Bl[32 * 136];

  int tid  = threadIdx.x;
  int arow = tid >> 2, aseg = (tid & 3) * 8;
  int srowA = tm * 64 + arow; if (srowA >= cnt) srowA = cnt - 1;
  int offs = offsets[e];
  const unsigned short* asrc0 = H + (long)(offs + srowA) * HID + aseg;
  unsigned short* adst = Al + arow * 40 + aseg;
  int brow = tid >> 3, bcol = (tid & 7) * 16;
  const float* wce = wc + (long)e * HID * EMB + (long)brow * EMB + bn * 128 + bcol;

  int lane = tid & 31;
  int wv   = tid >> 5;
  int mi0  = (wv & 1) * 2;
  int ni0  = (wv >> 1) * 2;

  v8f c00 = {}; v8f c01 = {}; v8f c10 = {}; v8f c11 = {};

  for (int k0 = 0; k0 < HID; k0 += 32) {
    async_cp16(adst, asrc0 + k0);
    {
      const float* pc = wce + (long)k0 * EMB;
      if (k0 + 32 < HID) __builtin_prefetch(pc + 32 * EMB, 0, 1);
      unsigned short* d = Bl + brow * 136 + bcol;
      *(uint4*)d       = cvt8(pc);
      *(uint4*)(d + 8) = cvt8(pc + 8);
    }
    wait_async0();
    __syncthreads();
    v16bf a0 = ldsA(Al, mi0,     lane);
    v16bf a1 = ldsA(Al, mi0 + 1, lane);
    v16bf b0 = ldsB(Bl, ni0,     lane, 136);
    v16bf b1 = ldsB(Bl, ni0 + 1, lane, 136);
    c00 = bfwmma(a0, b0, c00);
    c01 = bfwmma(a0, b1, c01);
    c10 = bfwmma(a1, b0, c10);
    c11 = bfwmma(a1, b1, c11);
    __syncthreads();
  }

  int hi   = lane >> 4;
  int col0 = bn * 128 + ni0 * 16 + (lane & 15);
  int col1 = col0 + 16;
#pragma unroll
  for (int m = 0; m < 2; m++) {
    v8f a0 = m ? c10 : c00, a1 = m ? c11 : c01;
#pragma unroll
    for (int r = 0; r < 8; r++) {
      int srow = tm * 64 + (mi0 + m) * 16 + r + 8 * hi;
      if (srow < cnt) {
        int   tok = tok_id[e * NTOK + srow];
        float g   = tok_gate[e * NTOK + srow];
        long  ob  = (long)tok * EMB;
        atomicAdd(&out[ob + col0], g * a0[r]);
        atomicAdd(&out[ob + col1], g * a1[r]);
      }
    }
  }
}

// ---------- host launcher ----------
extern "C" void kernel_launch(void* const* d_in, const int* in_sizes, int n_in,
                              void* d_out, int out_size, void* d_ws, size_t ws_size,
                              hipStream_t stream) {
  (void)in_sizes; (void)n_in; (void)out_size; (void)ws_size;
  const float* x  = (const float*)d_in[0];
  const float* wg = (const float*)d_in[1];
  const float* w1 = (const float*)d_in[2];
  const float* w2 = (const float*)d_in[3];
  const float* wc = (const float*)d_in[4];
  float* out = (float*)d_out;

  char* ws = (char*)d_ws;
  int*   counts   = (int*)ws;                                   // 16 ints
  int*   offsets  = (int*)(ws + 256);                           // 17 ints
  int*   tok_id   = (int*)(ws + 512);                           // 16*8192 ints
  float* tok_gate = (float*)(ws + 512 + (size_t)NE * NTOK * 4); // 16*8192 floats
  unsigned short* xb = (unsigned short*)(ws + 512 + (size_t)2 * NE * NTOK * 4); // 16 MB
  unsigned short* Hb = xb + (size_t)NTOK * EMB;                 // 32 MB

  {
    int threads = (NTOK * EMB) / 4;
    k_init<<<(threads + 255) / 256, 256, 0, stream>>>(x, xb, out, counts);
  }
  k_router<<<NTOK / 256, 256, 0, stream>>>(x, wg, counts, tok_id, tok_gate);
  k_offsets<<<1, 32, 0, stream>>>(counts, offsets);
  k_gemm1<<<dim3(HID / 128, NTOK / 64, NE), 256, 0, stream>>>(xb, w1, w2, counts,
                                                              offsets, tok_id, Hb);
  k_gemm2<<<dim3(EMB / 128, NTOK / 64, NE), 256, 0, stream>>>(Hb, wc, counts, offsets,
                                                              tok_id, tok_gate, out);
}
